// RNNTJointNetwork_83305185673308
// MI455X (gfx1250) — compile-verified, matrix operand order checked by reading, not run on hardware
//
#include <hip/hip_runtime.h>
#include <hip/hip_bf16.h>
#include <stdint.h>

#define B_    8
#define T_    256
#define U_    64
#define D_    512     // ENC_DIM == PRED_DIM == JOINT_DIM
#define V_    300
#define VPAD  304     // 19 * 16

#define A_SMEM_BYTES (8 * 16 * 32 * 16 * 2)   // 131072: 8 subtiles x 16 chunks x 1KB
#define B_ROW_BYTES  1040                     // 1024 data + 16 pad (TDM pad feature)
#define B_SLOT_BYTES (16 * B_ROW_BYTES)       // 16640 per wave
#define SMEM_TOTAL   (A_SMEM_BYTES + 8 * B_SLOT_BYTES)  // 264192 < 320KB

typedef __attribute__((ext_vector_type(16))) __bf16 v16bf;
typedef __attribute__((ext_vector_type(8)))  __bf16 v8bf;
typedef __attribute__((ext_vector_type(8)))  float  v8f;
typedef __attribute__((ext_vector_type(4)))  float  v4f;
typedef __attribute__((ext_vector_type(2)))  float  v2f;
typedef __attribute__((ext_vector_type(4)))  unsigned int u32x4;
typedef __attribute__((ext_vector_type(8)))  unsigned int u32x8;

__device__ __forceinline__ float fast_tanh(float x) {
#if __has_builtin(__builtin_amdgcn_tanhf)
  return __builtin_amdgcn_tanhf(x);
#else
  return tanhf(x);
#endif
}

// ---------------------------------------------------------------------------
// W_out (fp32, 300x512) -> bf16, padded to 304 rows (rows 300..303 = 0)
// ---------------------------------------------------------------------------
__global__ void cvt_wout_bf16(const float* __restrict__ w, __bf16* __restrict__ o) {
  int i = blockIdx.x * 256 + threadIdx.x;
  if (i >= VPAD * D_) return;
  int row = i >> 9;  // /512
  o[i] = (row < V_) ? (__bf16)w[i] : (__bf16)0.0f;
}

// ---------------------------------------------------------------------------
// Stage 1: Y[M,512] = X[M,512] @ W[512,512]^T + bias   (exact fp32 WMMA)
// ---------------------------------------------------------------------------
__global__ __launch_bounds__(256) void proj_f32_wmma(
    const float* __restrict__ X, const float* __restrict__ W,
    const float* __restrict__ bias, float* __restrict__ Y, int Mtiles) {
  const int lane   = threadIdx.x & 31;
  const int wid    = threadIdx.x >> 5;
  const int tile   = blockIdx.x * 8 + wid;
  const int NT     = D_ / 16;              // 32 N-tiles
  const int mt     = tile / NT;
  const int nt     = tile % NT;
  if (mt >= Mtiles) return;
  const int laneHi = lane >> 4;
  const int ln15   = lane & 15;
  const int m      = mt * 16 + ln15;
  const int j      = nt * 16 + ln15;

  const float* xp = X + (size_t)m * D_ + laneHi * 2;
  const float* wp = W + (size_t)j * D_ + laneHi * 2;

  v8f acc = {};
#pragma unroll 8
  for (int k = 0; k < D_; k += 4) {
    v2f a = *(const v2f*)(xp + k);
    v2f b = *(const v2f*)(wp + k);
    acc = __builtin_amdgcn_wmma_f32_16x16x4_f32(false, a, false, b,
                                                (short)0, acc, false, false);
  }
  const float bj = bias[j];
#pragma unroll
  for (int e = 0; e < 8; ++e) {
    const int r = e + laneHi * 8;
    Y[(size_t)(mt * 16 + r) * D_ + j] = acc[e] + bj;
  }
}

// ---------------------------------------------------------------------------
// Stage 2 (fused joint + vocab GEMM).
// Phase 1: tanh(enc+pred) -> bf16 A-frags, stored to LDS conflict-free
//          (lo at lane*16, hi at +512+lane*16: linear 64-bank sweep).
// Phase 2: per vocab tile, the wave's W_out 16x512 bf16 tile is DMA'd into
//          its private LDS slot by the Tensor Data Mover (tensor_load_to_lds,
//          D# built per ISA 8.3/8.4, pad_interval=7/pad_amount=3 => 1040B row
//          stride for conflict-free ds_load_b128), completion via
//          s_wait_tensorcnt; B-frags then lifted to registers and reused
//          across all 8 M-subtiles (W_out read once per workgroup).
// ---------------------------------------------------------------------------
__global__ __launch_bounds__(256) void rnnt_joint_vocab(
    const float* __restrict__ encp,    // [B*T, 512] fp32
    const float* __restrict__ predp,   // [B*U, 512] fp32
    const __bf16* __restrict__ woutb,  // [304, 512] bf16 (padded)
    const float* __restrict__ bout,    // [300]
    float* __restrict__ out) {         // [B*T*U, 300]
  extern __shared__ __align__(16) char smem_raw[];
  char* aSmem = smem_raw;                 // A fragments, 128 KB
  char* bSmem = smem_raw + A_SMEM_BYTES;  // TDM B slots, 8 x 16640 B

  const int lane   = threadIdx.x & 31;
  const int wid    = threadIdx.x >> 5;
  const int laneHi = lane >> 4;
  const int kh     = laneHi * 8;
  const int ln15   = lane & 15;

  // Warm L2 for the W_out tiles the TDM will fetch (global_prefetch_b8).
  {
    const char* wbase = (const char*)woutb;
    size_t off = ((size_t)threadIdx.x * (VPAD * D_ * 2 / 256));
    __builtin_prefetch(wbase + off, 0, 3);
  }

  // ---- Phase 1: build A fragments for this wave's 16-row subtile ----------
  const int sub = blockIdx.x * 8 + wid;     // 0..8191 M-subtiles
  const int m0  = sub << 4;
  const int bt  = m0 >> 6;                  // b*T + t  (U=64)
  const int u0  = m0 & (U_ - 1);
  const int b   = bt >> 8;                  // / T_

  const float* ep = encp  + (size_t)bt * D_ + kh;
  const float* pp = predp + (size_t)(b * U_ + u0 + ln15) * D_ + kh;

#pragma unroll
  for (int c = 0; c < 16; ++c) {
    const v4f* e4 = (const v4f*)(ep + c * 32);
    const v4f* p4 = (const v4f*)(pp + c * 32);
    float ev[16], pv[16];
    *(v4f*)(ev + 0)  = e4[0];  *(v4f*)(ev + 4)  = e4[1];   // K = kh..kh+7
    *(v4f*)(ev + 8)  = e4[4];  *(v4f*)(ev + 12) = e4[5];   // K = 16+kh..23+kh
    *(v4f*)(pv + 0)  = p4[0];  *(v4f*)(pv + 4)  = p4[1];
    *(v4f*)(pv + 8)  = p4[4];  *(v4f*)(pv + 12) = p4[5];
    v8bf lo, hi;
#pragma unroll
    for (int i = 0; i < 8; ++i) {
      lo[i] = (__bf16)fast_tanh(ev[i] + pv[i]);
      hi[i] = (__bf16)fast_tanh(ev[i + 8] + pv[i + 8]);
    }
    char* slot = aSmem + (size_t)(wid * 16 + c) * 1024;
    *(v8bf*)(slot + lane * 16)       = lo;   // conflict-free linear sweep
    *(v8bf*)(slot + 512 + lane * 16) = hi;
  }

  __syncthreads();

  // ---- Phase 2: vocab tiles; B via Tensor Data Mover ----------------------
  const int mblk = blockIdx.x * 128;
  const unsigned ldsSlot = (unsigned)A_SMEM_BYTES + (unsigned)wid * B_SLOT_BYTES;

  for (int nt = wid; nt < 19; nt += 8) {
    // --- Build Tensor DMA Descriptor (D#) and issue TDM load -------------
    {
      uint64_t ga = (uint64_t)(uintptr_t)woutb + (uint64_t)nt * 16 * D_ * 2;
      u32x4 g0;
      g0[0] = 1u;                                    // count=1 (valid user D#)
      g0[1] = ldsSlot;                               // lds_addr (bytes)
      g0[2] = (uint32_t)ga;                          // global_addr[31:0]
      g0[3] = (uint32_t)((ga >> 32) & 0x01FFFFFFu)   // global_addr[56:32]
            | (2u << 30);                            // type=2 ("image")
      u32x8 g1;
      g1[0] = (1u << 16)                             // data_size=1 -> 2 bytes
            | (1u << 20)                             // pad_enable
            | (7u << 22)                             // pad_interval: 256 DWORDs
            | (3u << 25);                            // pad_amount: 4 DWORDs
      g1[1] = (512u << 16);                          // tensor_dim0[15:0]=512
      g1[2] = (304u << 16);                          // dim0 hi=0 | tensor_dim1=304
      g1[3] = (512u << 16);                          // dim1 hi=0 | tile_dim0=512
      g1[4] = 16u;                                   // tile_dim1=16, tile_dim2=0
      g1[5] = 512u;                                  // tensor_dim0_stride lo
      g1[6] = 0u;                                    // stride0 hi | stride1 lo
      g1[7] = 0u;
      u32x4 g2 = {0u, 0u, 0u, 0u};                   // dims 2/3 unused
      u32x4 g3 = {0u, 0u, 0u, 0u};
      asm volatile("tensor_load_to_lds %0, %1, %2, %3"
                   :
                   : "s"(g0), "s"(g1), "s"(g2), "s"(g3)
                   : "memory");
      __builtin_amdgcn_s_wait_tensorcnt(0);
    }

    // --- Lift B fragments from LDS slot into registers -------------------
    v16bf bfr[16];
    const char* bp = bSmem + (size_t)(wid * B_SLOT_BYTES) + (size_t)ln15 * B_ROW_BYTES;
#pragma unroll
    for (int c = 0; c < 16; ++c) {
      v8bf blo = *(const v8bf*)(bp + (c * 32 + kh) * 2);
      v8bf bhi = *(const v8bf*)(bp + (c * 32 + kh) * 2 + 32);
      v16bf bb;
#pragma unroll
      for (int i = 0; i < 8; ++i) { bb[i] = blo[i]; bb[i + 8] = bhi[i]; }
      bfr[c] = bb;
    }

    const int v  = nt * 16 + ln15;
    const float bv = (v < V_) ? bout[v] : 0.0f;

    for (int s = 0; s < 8; ++s) {
      v8f acc = {};
#pragma unroll
      for (int c = 0; c < 16; ++c) {
        const char* ap = aSmem + (size_t)(s * 16 + c) * 1024;
        v8bf alo = *(const v8bf*)(ap + lane * 16);
        v8bf ahi = *(const v8bf*)(ap + 512 + lane * 16);
        v16bf aa;
#pragma unroll
        for (int i = 0; i < 8; ++i) { aa[i] = alo[i]; aa[i + 8] = ahi[i]; }
        acc = __builtin_amdgcn_wmma_f32_16x16x32_bf16(false, aa, false, bfr[c],
                                                      (short)0, acc, false, false);
      }
      if (v < V_) {
        const int r0 = mblk + s * 16;
#pragma unroll
        for (int e = 0; e < 8; ++e) {
          const int r = e + laneHi * 8;
          out[(size_t)(r0 + r) * V_ + v] = acc[e] + bv;
        }
      }
    }
  }
}

// ---------------------------------------------------------------------------
extern "C" void kernel_launch(void* const* d_in, const int* in_sizes, int n_in,
                              void* d_out, int out_size, void* d_ws, size_t ws_size,
                              hipStream_t stream) {
  (void)in_sizes; (void)n_in; (void)out_size; (void)ws_size;
  const float* enc    = (const float*)d_in[0];  // [8,256,512]
  const float* pred   = (const float*)d_in[1];  // [8,64,512]
  const float* W_enc  = (const float*)d_in[2];  // [512,512]
  const float* b_enc  = (const float*)d_in[3];  // [512]
  const float* W_pred = (const float*)d_in[4];  // [512,512]
  const float* b_pred = (const float*)d_in[5];  // [512]
  const float* W_out  = (const float*)d_in[6];  // [300,512]
  const float* b_out  = (const float*)d_in[7];  // [300]
  float* out = (float*)d_out;

  // Workspace carve-up (~5.6 MB total)
  float*  encp  = (float*)d_ws;                               // 2048*512 fp32
  float*  predp = encp + (size_t)(B_ * T_) * D_;              //  512*512 fp32
  __bf16* woutb = (__bf16*)(predp + (size_t)(B_ * U_) * D_);  //  304*512 bf16

  cvt_wout_bf16<<<(VPAD * D_ + 255) / 256, 256, 0, stream>>>(W_out, woutb);

  // enc_proj: 128 Mtiles x 32 Ntiles = 4096 wave-tiles / 8 = 512 blocks
  proj_f32_wmma<<<512, 256, 0, stream>>>(enc, W_enc, b_enc, encp, 128);
  // pred_proj: 32 x 32 = 1024 wave-tiles / 8 = 128 blocks
  proj_f32_wmma<<<128, 256, 0, stream>>>(pred, W_pred, b_pred, predp, 32);

  // vocab GEMM: 1024 workgroups x 8 waves, 258 KB dynamic LDS each
  rnnt_joint_vocab<<<1024, 256, SMEM_TOTAL, stream>>>(encp, predp, woutb,
                                                      b_out, out);
}
